// CliffordEPModel_61057255080245
// MI455X (gfx1250) — matrix operand order repaired and checked by reading, not compiled, not run on hardware
//
#include <hip/hip_runtime.h>

// ---------------------------------------------------------------------------
// CliffordEPModel on MI455X (gfx1250, wave32, WMMA)
//
// The whole reference network is linear in x:  pred = X(8192x512) @ Wtot(512x64)
// with Wtot folded from W_in, W_out, the Cl(3,0) Cayley signs and the EP
// relaxation scale (1 - 0.9^20).
//
// Kernel 1 (fold):  fp32, LDS-tiled contraction over H=512, writes Wtot
//                   directly in WMMA B-fragment layout as f16 hi + f16 lo.
// Kernel 2 (gemm):  wave-per-16-row-tile WMMA GEMM, f16 hi/lo split for
//                   fp32-grade accuracy, non-temporal streaming of X.
// ---------------------------------------------------------------------------

typedef __attribute__((ext_vector_type(16))) _Float16 v16h;
typedef __attribute__((ext_vector_type(8)))  float    v8f;
typedef __attribute__((ext_vector_type(4)))  float    v4f;   // clang vector (NT-load legal)

#define BATCH    8192
#define IN_DIM   64
#define HIDDEN   512
#define OUT_DIM  64
#define IB       8                    // 2^3 blades, Cl(3,0)
#define K_DIM    (IN_DIM * IB)        // 512  (GEMM K)
#define HK       (HIDDEN * IB)        // 4096
#define NFRAG    (K_DIM / 32 * 4 * 32 * 16)  // 32768 halves per hi/lo plane
#define EP_SCALE 0.87842334540943070f // 1 - 0.9^20

// Cayley sign for Cl(3,0): C[a,b,a^b] = (-1)^(reorder swaps); metric all +1.
__device__ __forceinline__ float csign(int a, int b) {
  int s = 0;
#pragma unroll
  for (int i = 0; i < 3; ++i)
    if ((b >> i) & 1) s += __popc(a >> (i + 1));
  return (s & 1) ? -1.0f : 1.0f;
}

// ---------------------------------------------------------------------------
// Fold kernel: Wtot[(m,alpha),o] = scale * sum_{c,h}
//     csign(alpha,c) * csign(alpha^c,alpha^c) * W_in[h,m,c] * W_out[o,h,alpha^c]
// grid = 64 (m), block = (o=64, alpha=8) = 512 threads.
// Output written as WMMA B fragments: idx(t,n,lane,j) with
//   K = t*32 + (lane>>4)*16 + j,  N = n*16 + (lane&15).
// ---------------------------------------------------------------------------
#define TILE_H 16
__global__ __launch_bounds__(512) void fold_kernel(
    const float* __restrict__ Win,    // (512, 64, 8)
    const float* __restrict__ Wout,   // (64, 512, 8)
    _Float16* __restrict__ bfrag)     // [NFRAG] hi, then [NFRAG] lo
{
  __shared__ float WoutS[OUT_DIM * (TILE_H * IB + 1)];  // stride 129 -> no bank conflicts
  __shared__ float WinS[TILE_H * IB];

  const int m     = blockIdx.x;
  const int o     = threadIdx.x;      // 0..63
  const int alpha = threadIdx.y;      // 0..7  (uniform per wave)
  const int tid   = threadIdx.y * 64 + threadIdx.x;

  // Per-thread sign table indexed by weight blade c (constant index only).
  float Sa[IB];
#pragma unroll
  for (int c = 0; c < IB; ++c) {
    const int a = alpha ^ c;          // Wx blade that pairs with W_out blade a
    Sa[c] = csign(alpha, c) * csign(a, a) * EP_SCALE;
  }

  float acc = 0.0f;
  for (int h0 = 0; h0 < HIDDEN; h0 += TILE_H) {
    __syncthreads();
    if (tid < TILE_H * IB)            // 128 floats of W_in[:, m, :]
      WinS[tid] = Win[(h0 + (tid >> 3)) * K_DIM + m * IB + (tid & 7)];
    for (int i = tid; i < OUT_DIM * TILE_H * IB; i += 512) {
      const int oo  = i >> 7;         // /128
      const int rem = i & 127;
      WoutS[oo * 129 + rem] = Wout[oo * HK + h0 * IB + rem];
    }
    __syncthreads();

#pragma unroll 4
    for (int hh = 0; hh < TILE_H; ++hh) {
      float w[IB];
#pragma unroll
      for (int c = 0; c < IB; ++c) w[c] = WinS[hh * IB + c];   // broadcast
      const float* wo = &WoutS[o * 129 + hh * IB];
#pragma unroll
      for (int c = 0; c < IB; ++c)
        acc = fmaf(Sa[c] * w[c], wo[alpha ^ c], acc);          // dynamic idx -> LDS, fine
    }
  }

  // Scatter into B-fragment layout (hi + lo split).
  const int K   = m * IB + alpha;     // 0..511
  const int t   = K >> 5;
  const int kr  = K & 31;
  const int ln  = (kr >> 4) * 16 + (o & 15);
  const int j   = kr & 15;
  const int n   = o >> 4;
  const int idx = (((t * 4 + n) * 32) + ln) * 16 + j;
  const _Float16 hi = (_Float16)acc;
  bfrag[idx]         = hi;
  bfrag[NFRAG + idx] = (_Float16)(acc - (float)hi);
}

// ---------------------------------------------------------------------------
// Main GEMM: out(8192x64) = X(8192x512) @ Wtot(512x64)
// One wave per 16-row tile; 4 column tiles (N=64); K = 16 steps of 32.
// A fragment per lane L: row = L%16, K = (L>>4)*16 + j (same k order as B).
// B fragments come pre-packed (L2-resident, 128 KB total).
// ---------------------------------------------------------------------------
#define WPB 4
__global__ __launch_bounds__(128) void gemm_kernel(
    const float* __restrict__ X,         // (8192, 512)
    const _Float16* __restrict__ bfrag,  // hi + lo fragment planes
    float* __restrict__ out)             // (8192, 64)
{
  const int lane  = threadIdx.x & 31;
  const int wid   = threadIdx.x >> 5;
  const int base8 = (lane >> 4) << 3;    // C row offset (0 or 8)
  const int col   = lane & 15;
  const int nRowTiles = BATCH / 16;      // 512

  for (int rt = blockIdx.x * WPB + wid; rt < nRowTiles; rt += gridDim.x * WPB) {
    const float* xrow = X + (size_t)(rt * 16 + col) * K_DIM;
    v8f acc[4] = {};

    for (int t = 0; t < K_DIM / 32; ++t) {
      // ---- A fragment: 16 fp32 -> f16 hi/lo in registers (stream X, NT) ----
      const float* xp = xrow + t * 32 + (lane >> 4) * 16;
      const v4f f0 = __builtin_nontemporal_load((const v4f*)(xp));
      const v4f f1 = __builtin_nontemporal_load((const v4f*)(xp + 4));
      const v4f f2 = __builtin_nontemporal_load((const v4f*)(xp + 8));
      const v4f f3 = __builtin_nontemporal_load((const v4f*)(xp + 12));
      const float tmp[16] = {f0[0], f0[1], f0[2], f0[3], f1[0], f1[1], f1[2], f1[3],
                             f2[0], f2[1], f2[2], f2[3], f3[0], f3[1], f3[2], f3[3]};
      v16h Ah, Al;
#pragma unroll
      for (int j = 0; j < 16; ++j) {
        const _Float16 h = (_Float16)tmp[j];
        Ah[j] = h;
        Al[j] = (_Float16)(tmp[j] - (float)h);
      }

      // ---- 4 N-tiles, hi/lo split: A*B = Ah*Bh + Al*Bh + Ah*Bl ----
#pragma unroll
      for (int n = 0; n < 4; ++n) {
        const v16h* bp = (const v16h*)bfrag + ((t * 4 + n) * 32 + lane);
        const v16h Bh = bp[0];
        const v16h Bl = bp[NFRAG / 16];
        acc[n] = __builtin_amdgcn_wmma_f32_16x16x32_f16(
            false, Ah, false, Bh, (short)0, acc[n], false, false);
        acc[n] = __builtin_amdgcn_wmma_f32_16x16x32_f16(
            false, Al, false, Bh, (short)0, acc[n], false, false);
        acc[n] = __builtin_amdgcn_wmma_f32_16x16x32_f16(
            false, Ah, false, Bl, (short)0, acc[n], false, false);
      }
    }

    // ---- store C: VGPR r -> row 8*(lane>>4)+r, col lane&15 ----
    const int rbase = rt * 16 + base8;   // base8 == 8*(lane>>4)
#pragma unroll
    for (int n = 0; n < 4; ++n) {
#pragma unroll
      for (int r = 0; r < 8; ++r) {
        __builtin_nontemporal_store(
            acc[n][r], &out[(size_t)(rbase + r) * OUT_DIM + n * 16 + col]);
      }
    }
  }
}

// ---------------------------------------------------------------------------
extern "C" void kernel_launch(void* const* d_in, const int* in_sizes, int n_in,
                              void* d_out, int out_size, void* d_ws, size_t ws_size,
                              hipStream_t stream) {
  (void)in_sizes; (void)n_in; (void)out_size; (void)ws_size;
  const float* x    = (const float*)d_in[0];   // (8192, 64, 8)
  const float* Win  = (const float*)d_in[1];   // (512, 64, 8)
  const float* Wout = (const float*)d_in[2];   // (64, 512, 8)
  float*     out    = (float*)d_out;           // (8192, 64)
  _Float16*  bfrag  = (_Float16*)d_ws;         // 2 * 64 KB fragment planes

  fold_kernel<<<dim3(IN_DIM), dim3(64, 8), 0, stream>>>(Win, Wout, bfrag);
  gemm_kernel<<<dim3(128), dim3(128), 0, stream>>>(x, bfrag, out);
}
